// SparseOffsetDict_24180665876974
// MI455X (gfx1250) — matrix operand order repaired and checked by reading, not compiled, not run on hardware
//
#include <hip/hip_runtime.h>
#include <stdint.h>

// ---- types (ext-vector: trivially constructible, union-safe) ----
typedef __attribute__((ext_vector_type(16))) __bf16        v16bf;
typedef __attribute__((ext_vector_type(8)))  float         v8f;
typedef __attribute__((ext_vector_type(4)))  unsigned int  v4u;
typedef __attribute__((ext_vector_type(4)))  float         v4f;
typedef __attribute__((ext_vector_type(4)))  unsigned short v4h;

union Frag { v4u u[2]; v16bf v; };   // 32 bytes = one 16-bit WMMA operand per lane

// ---- problem constants ----
#define NTOK   8192      // B*T
#define DDIM   1024      // D
#define MDICT  16384     // M
#define TOPK   8

// ---- tiling ----
#define TM 128           // tokens per workgroup tile
#define TN 128           // dict units per GEMM sub-tile
#define KB 64            // K slab staged through LDS
#define NSLAB (DDIM / KB)              // 16
#define NSTRIPE 2048     // dict units owned per workgroup (looped in TN chunks)
#define NSUB (NSTRIPE / TN)            // 16
#define NSTRIPES (MDICT / NSTRIPE)     // 8
#define CAND_PER_STRIPE 16             // 2 half-scans x top-8
#define CANDS (NSTRIPES * CAND_PER_STRIPE)  // 128 candidates per token

__device__ __forceinline__ unsigned short f2bf(float f) {
    unsigned u = __float_as_uint(f);
    unsigned r = u + 0x7FFFu + ((u >> 16) & 1u);   // round-to-nearest-even
    return (unsigned short)(r >> 16);
}

// CDNA5 async copy: global -> LDS, 16 bytes per lane, tracked by ASYNCcnt.
__device__ __forceinline__ void async_copy16(unsigned lds_off, const void* g) {
    asm volatile("global_load_async_to_lds_b128 %0, %1, off"
                 :: "v"(lds_off), "v"((unsigned long long)(uintptr_t)g)
                 : "memory");
}
__device__ __forceinline__ void wait_async0() {
#if __has_builtin(__builtin_amdgcn_s_wait_asynccnt)
    __builtin_amdgcn_s_wait_asynccnt(0);
#else
    asm volatile("s_wait_asynccnt 0x0" ::: "memory");
#endif
}

// ---------------------------------------------------------------------------
// Kernel 1: fp32 -> bf16 conversion of x and W_enc; also zero the loss scalar.
// ---------------------------------------------------------------------------
__global__ void convert_bf16_kernel(const float* __restrict__ x,
                                    const float* __restrict__ w,
                                    unsigned short* __restrict__ xb,
                                    unsigned short* __restrict__ wb,
                                    float* __restrict__ loss) {
    if (blockIdx.x == 0 && threadIdx.x == 0) *loss = 0.0f;
    const long n1 = (long)NTOK * DDIM / 4;
    const long n2 = (long)MDICT * DDIM / 4;
    long i = (long)blockIdx.x * blockDim.x + threadIdx.x;
    long stride = (long)gridDim.x * blockDim.x;
    for (long j = i; j < n1; j += stride) {
        v4f f = ((const v4f*)x)[j];
        v4h o; o.x = f2bf(f.x); o.y = f2bf(f.y); o.z = f2bf(f.z); o.w = f2bf(f.w);
        ((v4h*)xb)[j] = o;
    }
    for (long j = i; j < n2; j += stride) {
        v4f f = ((const v4f*)w)[j];
        v4h o; o.x = f2bf(f.x); o.y = f2bf(f.y); o.z = f2bf(f.z); o.w = f2bf(f.w);
        ((v4h*)wb)[j] = o;
    }
}

// ---------------------------------------------------------------------------
// Kernel 2: fused bf16-WMMA encode GEMM + streaming per-stripe top-8.
// Async-to-LDS double-buffered pipeline; 8 waves: 2 m-groups x 4 n-groups.
// grid = (NTOK/TM, MDICT/NSTRIPE), block = 256.
// ---------------------------------------------------------------------------
__global__ __launch_bounds__(256)
void encode_topk_kernel(const unsigned short* __restrict__ xb,
                        const unsigned short* __restrict__ wb,
                        float* __restrict__ cand_val,
                        int*   __restrict__ cand_idx) {
    __shared__ __align__(16) unsigned short Asl[2][TM][KB + 8];  // 144B row stride
    __shared__ __align__(16) unsigned short Bsl[2][TN][KB + 8];
    __shared__ float Csc[TM][TN + 5];                            // 133: conflict-free

    const int tid  = threadIdx.x;
    const int lane = tid & 31;
    const int wid  = tid >> 5;
    const int mg   = wid & 1;        // m-group: 0..1 (64 tokens each)
    const int ng   = wid >> 1;       // n-group: 0..3 (32 dict units each)
    const int l16  = lane & 15;
    const int hhi  = lane >> 4;      // half-wave select

    const int tokBase     = blockIdx.x * TM;
    const int nStripeBase = blockIdx.y * NSTRIPE;

    // fixed per-thread staging geometry: 4 x 16B chunks per matrix per slab
    int rr_[4], cc_[4];
    unsigned aOff[2][4], bOff[2][4];
    const unsigned short* aG[4];
    #pragma unroll
    for (int i = 0; i < 4; ++i) {
        const int ch = i * 256 + tid;
        rr_[i] = ch >> 3; cc_[i] = ch & 7;
        aOff[0][i] = (unsigned)(unsigned long long)(void*)&Asl[0][rr_[i]][cc_[i] * 8];
        aOff[1][i] = (unsigned)(unsigned long long)(void*)&Asl[1][rr_[i]][cc_[i] * 8];
        bOff[0][i] = (unsigned)(unsigned long long)(void*)&Bsl[0][rr_[i]][cc_[i] * 8];
        bOff[1][i] = (unsigned)(unsigned long long)(void*)&Bsl[1][rr_[i]][cc_[i] * 8];
        aG[i] = xb + (long)(tokBase + rr_[i]) * DDIM + cc_[i] * 8;
    }

    // per-thread running top-8 (statically indexed -> stays in VGPRs)
    float tv[TOPK]; int ti[TOPK];
    #pragma unroll
    for (int k = 0; k < TOPK; ++k) { tv[k] = -3.0e38f; ti[k] = 0; }
    const int tt = tid & (TM - 1);   // token row this thread scans
    const int hh = tid >> 7;         // which 64-column half it scans

    for (int ns = 0; ns < NSUB; ++ns) {
        const int nBase = nStripeBase + ns * TN;

        v8f c[4][2] = {};            // 8 accum tiles of 16x16 f32 per wave

        // issue K-slab 0 into buffer 0 (async engine fills LDS)
        #pragma unroll
        for (int i = 0; i < 4; ++i) {
            async_copy16(aOff[0][i], aG[i]);
            async_copy16(bOff[0][i], wb + (long)(nBase + rr_[i]) * DDIM + cc_[i] * 8);
        }
        wait_async0();
        __syncthreads();

        for (int ks = 0; ks < NSLAB; ++ks) {
            const int cur = ks & 1;
            if (ks + 1 < NSLAB) {        // overlap next slab's DMA with compute
                const int kn = (ks + 1) * KB;
                const int nb = 1 - cur;  // buffer last consumed at ks-1 (barrier-safe)
                #pragma unroll
                for (int i = 0; i < 4; ++i) {
                    async_copy16(aOff[nb][i], aG[i] + kn);
                    async_copy16(bOff[nb][i],
                                 wb + (long)(nBase + rr_[i]) * DDIM + kn + cc_[i] * 8);
                }
            }
            #pragma unroll
            for (int k0 = 0; k0 < KB; k0 += 32) {
                // A (16x32 bf16, MxK): lane holds M=l16, K = k0 + hhi*8 + {0..7,16..23}
                Frag af[4];
                #pragma unroll
                for (int a = 0; a < 4; ++a) {
                    const unsigned short* p = &Asl[cur][mg * 64 + a * 16 + l16][k0 + hhi * 8];
                    af[a].u[0] = *(const v4u*)p;
                    af[a].u[1] = *(const v4u*)(p + 16);
                }
                // B (32x16 bf16, KxN): lane holds N=l16, K = k0 + hhi*8 + {0..7,16..23}
                Frag bfr[2];
                #pragma unroll
                for (int b = 0; b < 2; ++b) {
                    const unsigned short* p = &Bsl[cur][ng * 32 + b * 16 + l16][k0 + hhi * 8];
                    bfr[b].u[0] = *(const v4u*)p;
                    bfr[b].u[1] = *(const v4u*)(p + 16);
                }
                #pragma unroll
                for (int a = 0; a < 4; ++a)
                    #pragma unroll
                    for (int b = 0; b < 2; ++b)
                        c[a][b] = __builtin_amdgcn_wmma_f32_16x16x32_bf16(
                            false, af[a].v, false, bfr[b].v,
                            (short)0, c[a][b], false, false);
            }
            if (ks + 1 < NSLAB) {
                wait_async0();           // own slab-(ks+1) writes landed in LDS
                __syncthreads();         // everyone's landed -> consume next iter
            }
        }
        // spill C tiles: VGPR r -> M = base + hhi*8 + r, N = base + l16
        #pragma unroll
        for (int a = 0; a < 4; ++a) {
            #pragma unroll
            for (int b = 0; b < 2; ++b) {
                const int m = mg * 64 + a * 16 + hhi * 8;
                const int n = ng * 32 + b * 16 + l16;
                #pragma unroll
                for (int r = 0; r < 8; ++r)
                    Csc[m + r][n] = c[a][b][r];
            }
        }
        __syncthreads();
        // streaming top-8: thread scans 64 columns of its token row
        for (int j = 0; j < 64; ++j) {
            float v = Csc[tt][hh * 64 + j];
            if (v > tv[TOPK - 1]) {
                tv[TOPK - 1] = v; ti[TOPK - 1] = nBase + hh * 64 + j;
                #pragma unroll
                for (int p = TOPK - 1; p > 0; --p) {
                    if (tv[p] > tv[p - 1]) {
                        float fv_ = tv[p]; tv[p] = tv[p - 1]; tv[p - 1] = fv_;
                        int   fi_ = ti[p]; ti[p] = ti[p - 1]; ti[p - 1] = fi_;
                    }
                }
            }
        }
        __syncthreads();   // scan done before next ns touches Csc / LDS buffers
    }

    // emit 16 candidates per (token, stripe): true stripe top-8 is a subset
    const long base = (long)(tokBase + tt) * CANDS
                    + (long)blockIdx.y * CAND_PER_STRIPE + hh * TOPK;
    #pragma unroll
    for (int k = 0; k < TOPK; ++k) {
        cand_val[base + k] = tv[k];
        cand_idx[base + k] = ti[k];
    }
}

// ---------------------------------------------------------------------------
// Kernel 3: merge 128 candidates -> final top-8, loss atomicAdd, gather-decode.
// grid = NTOK blocks of 256.
// ---------------------------------------------------------------------------
__global__ __launch_bounds__(256)
void merge_decode_kernel(const float* __restrict__ cand_val,
                         const int*   __restrict__ cand_idx,
                         const float* __restrict__ dict,
                         float* __restrict__ out,
                         float* __restrict__ loss) {
    __shared__ float sv[CANDS];
    __shared__ int   si[CANDS];
    __shared__ float fv[TOPK];
    __shared__ int   fi[TOPK];
    const int tok = blockIdx.x;
    const int tid = threadIdx.x;
    if (tid < CANDS) {
        sv[tid] = cand_val[(long)tok * CANDS + tid];
        si[tid] = cand_idx[(long)tok * CANDS + tid];
    }
    __syncthreads();
    if (tid == 0) {
        float asum = 0.0f;
        for (int k = 0; k < TOPK; ++k) {
            int mx = 0; float mv = sv[0];
            for (int j = 1; j < CANDS; ++j)
                if (sv[j] > mv) { mv = sv[j]; mx = j; }
            fv[k] = mv; fi[k] = si[mx]; sv[mx] = -3.4e38f;
            asum += fabsf(mv);
        }
        atomicAdd(loss, asum * (1.0f / ((float)NTOK * (float)MDICT)));
    }
    __syncthreads();
    float vv[TOPK]; long rr[TOPK];
    #pragma unroll
    for (int k = 0; k < TOPK; ++k) { vv[k] = fv[k]; rr[k] = (long)fi[k] * DDIM; }
    for (int d = tid; d < DDIM; d += 256) {
        float acc = 0.0f;
        #pragma unroll
        for (int k = 0; k < TOPK; ++k) acc += vv[k] * dict[rr[k] + d];
        out[(long)tok * DDIM + d] = acc;
    }
}

// ---------------------------------------------------------------------------
// Launch: convert -> encode+topk -> merge+decode.  ws usage: 56 MB
//   [ 0,16MB)  x bf16      [16,48MB)  W_enc bf16
//   [48,52MB)  cand_val    [52,56MB)  cand_idx
// ---------------------------------------------------------------------------
extern "C" void kernel_launch(void* const* d_in, const int* in_sizes, int n_in,
                              void* d_out, int out_size, void* d_ws, size_t ws_size,
                              hipStream_t stream) {
    (void)in_sizes; (void)n_in; (void)out_size; (void)ws_size;
    const float* x    = (const float*)d_in[0];
    const float* Wenc = (const float*)d_in[1];
    const float* dict = (const float*)d_in[2];
    float* out  = (float*)d_out;
    float* loss = out + (long)NTOK * DDIM;

    char* ws = (char*)d_ws;
    unsigned short* xb = (unsigned short*)ws;
    unsigned short* wb = (unsigned short*)(ws + (size_t)NTOK * DDIM * 2);
    float* cv = (float*)(ws + (size_t)NTOK * DDIM * 2 + (size_t)MDICT * DDIM * 2);
    int*   ci = (int*)((char*)cv + (size_t)NTOK * CANDS * sizeof(float));

    convert_bf16_kernel<<<2048, 256, 0, stream>>>(x, Wenc, xb, wb, loss);
    dim3 g(NTOK / TM, MDICT / NSTRIPE);
    encode_topk_kernel<<<g, 256, 0, stream>>>(xb, wb, cv, ci);
    merge_decode_kernel<<<NTOK, 256, 0, stream>>>(cv, ci, dict, out, loss);
}
